// FAGCNEncoder_75814762709163
// MI455X (gfx1250) — matrix operand order repaired and checked by reading, not compiled, not run on hardware
//
#include <hip/hip_runtime.h>

// ---------------------------------------------------------------------------
// FAGCN encoder for MI455X (gfx1250, wave32).
//   h   = ELU(X @ W1^T + b1)                  -> WMMA f16 (f32 accum), W1 via TDM
//   a_l = h @ att_l_w + b ; a_r = h @ att_r_w + b
//   deg/dinv, alpha = tanh(a_l[s]+a_r[d])
//   agg = scatter_add(dinv[s]*dinv[d]*alpha * h[s])
//   out = softmax((agg + 0.2 h) @ W2^T + b2)  -> WMMA f16 (f32 accum), W2 via TDM
// ---------------------------------------------------------------------------

typedef __attribute__((ext_vector_type(16))) _Float16     v16h;
typedef __attribute__((ext_vector_type(8)))  float        v8f;
typedef __attribute__((ext_vector_type(4)))  unsigned int u32x4;
typedef __attribute__((ext_vector_type(8)))  int          i32x8;
typedef __attribute__((ext_vector_type(4)))  int          i32x4;

#define N_NODES 100000
#define N_EDGES 1600000
#define F_DIM   128
#define H_DIM   64
#define K_OUT   16
#define EPS_F   0.2f

#if defined(__AMDGCN__) && __has_builtin(__builtin_amdgcn_tensor_load_to_lds)
#define USE_TDM 1
#else
#define USE_TDM 0
#endif

#if USE_TDM
// Issue a TDM DMA of a 2-D row-major f32 tile [dim1 rows x dim0 cols]
// (global -> LDS), then drain TENSORcnt. Caller: one wave only, then
// __syncthreads() in all waves. D# layout per CDNA5 ISA.
__device__ __forceinline__ void tdm_load_2d_f32(const float* gptr,
                                                unsigned lds_off,
                                                int dim0, int dim1) {
  const unsigned long long ga = (unsigned long long)(uintptr_t)gptr;
  u32x4 g0 = {0u, 0u, 0u, 0u};
  g0.s0 = 1u;                                        // count=1, user D#
  g0.s1 = lds_off;                                   // lds_addr (bytes)
  g0.s2 = (unsigned)(ga & 0xFFFFFFFFu);              // global_addr[31:0]
  g0.s3 = (unsigned)((ga >> 32) & 0x01FFFFFFu)       // global_addr[56:32]
          | 0x80000000u;                             // type=2 ("image")
  i32x8 g1 = {0, 0, 0, 0, 0, 0, 0, 0};
  g1.s0 = (2 << 16);                                 // data_size = 4 bytes
  g1.s1 = (dim0 & 0xFFFF) << 16;                     // tensor_dim0[15:0]
  g1.s2 = (dim1 & 0xFFFF) << 16;                     // tensor_dim1[15:0]
  g1.s3 = (dim0 & 0xFFFF) << 16;                     // tile_dim0
  g1.s4 = (dim1 & 0xFFFF);                           // tile_dim1 (tile_dim2=0)
  g1.s5 = dim0;                                      // tensor_dim0_stride[31:0]
  i32x4 gz = {0, 0, 0, 0};
#if __clang_major__ >= 23
  i32x8 gz8 = {0, 0, 0, 0, 0, 0, 0, 0};
  __builtin_amdgcn_tensor_load_to_lds(g0, g1, gz, gz, gz8, 0);
#else
  __builtin_amdgcn_tensor_load_to_lds(g0, g1, gz, gz, 0);
#endif
  __builtin_amdgcn_s_wait_tensorcnt(0);
}
#endif

// ------------------------------- kernel 1 ----------------------------------
// One wave computes a 16-node x 64-col strip of h. W1 staged in LDS via TDM,
// then converted once to f16 for the WMMA B fragments.
__global__ __launch_bounds__(256) void lin1_wmma_kernel(
    const float* __restrict__ X, const float* __restrict__ W1,
    const float* __restrict__ b1, float* __restrict__ h) {
  __shared__ float    sWf[H_DIM * F_DIM];  // 32 KB (TDM destination)
  __shared__ _Float16 sW[H_DIM * F_DIM];   // 16 KB (WMMA operand)
  const int tid = threadIdx.x;
#if USE_TDM
  if (tid < 32) {  // wave 0 drives the DMA; TENSORcnt drained before sync
    tdm_load_2d_f32(W1, (unsigned)(uintptr_t)&sWf[0], F_DIM, H_DIM);
  }
  __syncthreads();
  for (int i = tid; i < H_DIM * F_DIM; i += 256) sW[i] = (_Float16)sWf[i];
#else
  for (int i = tid; i < H_DIM * F_DIM; i += 256) sW[i] = (_Float16)W1[i];
#endif
  __syncthreads();

  const int wave = blockIdx.x * 8 + (tid >> 5);
  if (wave >= N_NODES / 16) return;
  const int lane = tid & 31;
  const int lh   = lane >> 4;   // half-wave: 0 or 1
  const int sub  = lane & 15;
  const int m0   = wave * 16;

  const float* __restrict__ xrow = X + (size_t)(m0 + sub) * F_DIM;

  v8f acc[4];
  acc[0] = v8f{}; acc[1] = v8f{}; acc[2] = v8f{}; acc[3] = v8f{};

#pragma unroll
  for (int kb = 0; kb < F_DIM; kb += 32) {
    // ---- A fragment: 16 f16 per lane, two contiguous 8-float runs ----
    v16h a;
    {
      const float4 p0 = *(const float4*)(xrow + kb + 8 * lh);
      const float4 p1 = *(const float4*)(xrow + kb + 8 * lh + 4);
      const float4 p2 = *(const float4*)(xrow + kb + 16 + 8 * lh);
      const float4 p3 = *(const float4*)(xrow + kb + 16 + 8 * lh + 4);
      a[0] = (_Float16)p0.x;  a[1] = (_Float16)p0.y;
      a[2] = (_Float16)p0.z;  a[3] = (_Float16)p0.w;
      a[4] = (_Float16)p1.x;  a[5] = (_Float16)p1.y;
      a[6] = (_Float16)p1.z;  a[7] = (_Float16)p1.w;
      a[8] = (_Float16)p2.x;  a[9] = (_Float16)p2.y;
      a[10] = (_Float16)p2.z; a[11] = (_Float16)p2.w;
      a[12] = (_Float16)p3.x; a[13] = (_Float16)p3.y;
      a[14] = (_Float16)p3.z; a[15] = (_Float16)p3.w;
    }
    // ---- 4 B fragments (cols 16t..16t+15) from LDS, accumulate ----
#pragma unroll
    for (int t = 0; t < 4; ++t) {
      const _Float16* wp = &sW[(size_t)(t * 16 + sub) * F_DIM + kb + 8 * lh];
      v16h b;
#pragma unroll
      for (int j = 0; j < 16; ++j) b[j] = wp[j < 8 ? j : j + 8];
      acc[t] = __builtin_amdgcn_wmma_f32_16x16x32_f16(
          false, a, false, b, (short)0, acc[t], false, false);
    }
  }

  // ---- epilogue: + b1, ELU, store f32 h ----
#pragma unroll
  for (int t = 0; t < 4; ++t) {
    const int col = t * 16 + sub;
    const float bias = b1[col];
#pragma unroll
    for (int r = 0; r < 8; ++r) {
      const int orow = m0 + r + 8 * lh;
      float v = acc[t][r] + bias;
      v = v > 0.f ? v : expm1f(v);
      h[(size_t)orow * H_DIM + col] = v;
    }
  }
}

// ------------------------------- kernel 2 ----------------------------------
__global__ void zero_f32_kernel(float* __restrict__ p, int n) {
  int i = blockIdx.x * blockDim.x + threadIdx.x;
  const int stride = gridDim.x * blockDim.x;
  for (; i < n; i += stride) p[i] = 0.f;
}

// ------------------------------- kernel 3 ----------------------------------
// Wave-per-node dual dot product: a_l, a_r.
__global__ __launch_bounds__(256) void attn_kernel(
    const float* __restrict__ h,
    const float* __restrict__ alw, const float* __restrict__ alb,
    const float* __restrict__ arw, const float* __restrict__ arb,
    float* __restrict__ a_l, float* __restrict__ a_r) {
  const int node = blockIdx.x * 8 + (threadIdx.x >> 5);
  if (node >= N_NODES) return;
  const int lane = threadIdx.x & 31;
  const float h0 = h[(size_t)node * H_DIM + lane];
  const float h1 = h[(size_t)node * H_DIM + lane + 32];
  float sl = h0 * alw[lane] + h1 * alw[lane + 32];
  float sr = h0 * arw[lane] + h1 * arw[lane + 32];
#pragma unroll
  for (int m = 16; m >= 1; m >>= 1) {
    sl += __shfl_xor(sl, m, 32);
    sr += __shfl_xor(sr, m, 32);
  }
  if (lane == 0) {
    a_l[node] = sl + alb[0];
    a_r[node] = sr + arb[0];
  }
}

// ------------------------------- kernel 4 ----------------------------------
__global__ void degree_kernel(const int* __restrict__ dst,
                              float* __restrict__ deg) {
  int i = blockIdx.x * blockDim.x + threadIdx.x;
  const int stride = gridDim.x * blockDim.x;
  for (; i < N_EDGES; i += stride) atomicAdd(&deg[dst[i]], 1.0f);
}

// ------------------------------- kernel 5 ----------------------------------
__global__ void dinv_kernel(const float* __restrict__ deg,
                            float* __restrict__ dinv) {
  int i = blockIdx.x * blockDim.x + threadIdx.x;
  const int stride = gridDim.x * blockDim.x;
  for (; i < N_NODES; i += stride) {
    const float d = deg[i];
    dinv[i] = d > 0.f ? rsqrtf(d) : 0.f;
  }
}

// ------------------------------- kernel 6 ----------------------------------
// Wave-per-edge: lanes span the 64 columns (coalesced gather + scatter).
__global__ __launch_bounds__(256) void scatter_kernel(
    const int* __restrict__ src, const int* __restrict__ dst,
    const float* __restrict__ h, const float* __restrict__ dinv,
    const float* __restrict__ a_l, const float* __restrict__ a_r,
    float* __restrict__ agg) {
  const int lane = threadIdx.x & 31;
  const int wid  = blockIdx.x * 8 + (threadIdx.x >> 5);
  const int nw   = gridDim.x * 8;
  for (int e = wid; e < N_EDGES; e += nw) {
    const int s = src[e];
    const int d = dst[e];
    const int en = e + nw;
    if (en < N_EDGES) {  // prefetch next source row (global_prefetch_b8)
      __builtin_prefetch(&h[(size_t)src[en] * H_DIM + lane], 0, 1);
    }
    const float coeff = dinv[s] * dinv[d] * tanhf(a_l[s] + a_r[d]);
    const float v0 = h[(size_t)s * H_DIM + lane];
    const float v1 = h[(size_t)s * H_DIM + lane + 32];
    atomicAdd(&agg[(size_t)d * H_DIM + lane],      coeff * v0);
    atomicAdd(&agg[(size_t)d * H_DIM + lane + 32], coeff * v1);
  }
}

// ------------------------------- kernel 7 ----------------------------------
// Wave computes 16 nodes x 16 clusters: (agg + 0.2h) @ W2^T + b2, row softmax.
__global__ __launch_bounds__(256) void lin2_softmax_wmma_kernel(
    const float* __restrict__ agg, const float* __restrict__ h,
    const float* __restrict__ W2, const float* __restrict__ b2,
    float* __restrict__ out) {
  __shared__ float    sWf[K_OUT * H_DIM];  // 4 KB (TDM destination)
  __shared__ _Float16 sW[K_OUT * H_DIM];   // 2 KB (WMMA operand)
  const int tid = threadIdx.x;
#if USE_TDM
  if (tid < 32) {
    tdm_load_2d_f32(W2, (unsigned)(uintptr_t)&sWf[0], H_DIM, K_OUT);
  }
  __syncthreads();
  for (int i = tid; i < K_OUT * H_DIM; i += 256) sW[i] = (_Float16)sWf[i];
#else
  for (int i = tid; i < K_OUT * H_DIM; i += 256) sW[i] = (_Float16)W2[i];
#endif
  __syncthreads();

  const int wave = blockIdx.x * 8 + (tid >> 5);
  if (wave >= N_NODES / 16) return;
  const int lane = tid & 31;
  const int lh   = lane >> 4;
  const int sub  = lane & 15;
  const int m0   = wave * 16;

  const float* __restrict__ grow = agg + (size_t)(m0 + sub) * H_DIM;
  const float* __restrict__ hrow = h   + (size_t)(m0 + sub) * H_DIM;

  v8f acc = {};
#pragma unroll
  for (int kb = 0; kb < H_DIM; kb += 32) {
    v16h a;
#pragma unroll
    for (int j = 0; j < 8; ++j) {
      const int k = kb + 8 * lh + j;
      a[j] = (_Float16)(grow[k] + EPS_F * hrow[k]);
    }
#pragma unroll
    for (int j = 0; j < 8; ++j) {
      const int k = kb + 16 + 8 * lh + j;
      a[8 + j] = (_Float16)(grow[k] + EPS_F * hrow[k]);
    }
    v16h b;
    const _Float16* wp = &sW[(size_t)sub * H_DIM + kb + 8 * lh];
#pragma unroll
    for (int j = 0; j < 16; ++j) b[j] = wp[j < 8 ? j : j + 8];
    acc = __builtin_amdgcn_wmma_f32_16x16x32_f16(
        false, a, false, b, (short)0, acc, false, false);
  }

  const float bias = b2[sub];
  float vals[8];
#pragma unroll
  for (int r = 0; r < 8; ++r) vals[r] = acc[r] + bias;

  // softmax across the 16 clusters of each row (16-lane group reduction)
#pragma unroll
  for (int r = 0; r < 8; ++r) {
    float mx = vals[r];
#pragma unroll
    for (int x = 8; x >= 1; x >>= 1) mx = fmaxf(mx, __shfl_xor(mx, x, 32));
    const float ex = expf(vals[r] - mx);
    float sm = ex;
#pragma unroll
    for (int x = 8; x >= 1; x >>= 1) sm += __shfl_xor(sm, x, 32);
    out[(size_t)(m0 + r + 8 * lh) * K_OUT + sub] = ex / sm;
  }
}

// ------------------------------- launch ------------------------------------
extern "C" void kernel_launch(void* const* d_in, const int* in_sizes, int n_in,
                              void* d_out, int out_size, void* d_ws,
                              size_t ws_size, hipStream_t stream) {
  (void)in_sizes; (void)n_in; (void)out_size; (void)ws_size;

  const float* X   = (const float*)d_in[0];
  const int*   ei  = (const int*)d_in[1];
  // d_in[2] = batch (unused: single graph)
  const float* W1  = (const float*)d_in[3];
  const float* b1  = (const float*)d_in[4];
  const float* alw = (const float*)d_in[5];
  const float* alb = (const float*)d_in[6];
  const float* arw = (const float*)d_in[7];
  const float* arb = (const float*)d_in[8];
  const float* W2  = (const float*)d_in[9];
  const float* b2  = (const float*)d_in[10];
  float* out = (float*)d_out;

  const int* src = ei;            // edge_index[0]
  const int* dst = ei + N_EDGES;  // edge_index[1]

  // workspace layout (f32): h | agg | deg | dinv | a_l | a_r  (~53 MB)
  float* ws   = (float*)d_ws;
  float* h    = ws;
  float* agg  = ws + (size_t)N_NODES * H_DIM;
  float* deg  = ws + 2 * (size_t)N_NODES * H_DIM;
  float* dinv = deg + N_NODES;
  float* a_l  = dinv + N_NODES;
  float* a_r  = a_l + N_NODES;

  const int strips = N_NODES / 16;                 // 6250 waves
  const int gemm_blocks = (strips + 7) / 8;        // 8 waves / block

  lin1_wmma_kernel<<<gemm_blocks, 256, 0, stream>>>(X, W1, b1, h);
  // agg (N*H) and deg (N) are contiguous -> one clear
  zero_f32_kernel<<<1024, 256, 0, stream>>>(agg, N_NODES * H_DIM + N_NODES);
  attn_kernel<<<N_NODES / 8, 256, 0, stream>>>(h, alw, alb, arw, arb, a_l, a_r);
  degree_kernel<<<1024, 256, 0, stream>>>(dst, deg);
  dinv_kernel<<<512, 256, 0, stream>>>(deg, dinv);
  scatter_kernel<<<2048, 256, 0, stream>>>(src, dst, h, dinv, a_l, a_r, agg);
  lin2_softmax_wmma_kernel<<<gemm_blocks, 256, 0, stream>>>(agg, h, W2, b2, out);
}
